// Net_49563922596744
// MI455X (gfx1250) — matrix operand order repaired and checked by reading, not compiled, not run on hardware
//
#include <hip/hip_runtime.h>
#include <math.h>

typedef __attribute__((ext_vector_type(16))) __bf16 v16bf;
typedef __attribute__((ext_vector_type(8)))  __bf16 v8bf;
typedef __attribute__((ext_vector_type(8)))  float  v8f;

#ifndef M_PI_F
#define M_PI_F 3.14159265358979323846f
#endif

static __device__ __forceinline__ v8f wmma_bf16(v16bf a, v16bf b, v8f c) {
    // D = A(16x32 bf16) * B(32x16 bf16) + C(16x16 f32)
    return __builtin_amdgcn_wmma_f32_16x16x32_bf16(false, a, false, b,
                                                   (short)0, c, false, false);
}

static __device__ __forceinline__ v16bf cat8(v8bf lo, v8bf hi) {
    return __builtin_shufflevector(lo, hi, 0, 1, 2, 3, 4, 5, 6, 7,
                                           8, 9, 10, 11, 12, 13, 14, 15);
}

// ---------------------------------------------------------------------------
// conv3x3, pad=1, Cin=Cout=32, implicit GEMM via WMMA bf16.
// grid: x = ceil(W/32), y = H, z = B ; block = 128 (4 waves)
// Input tile staged in LDS channel-contiguous -> A frag = 2x ds_load_b128.
// Weights staged [tap][n][k] -> B frag = 2x ds_load_b128, same k order as A.
// optional fused leaky-relu (negslope>0) and residual add.
// ---------------------------------------------------------------------------
__global__ __launch_bounds__(128)
void k_conv3x3_wmma(const float* __restrict__ in, const float* __restrict__ w,
                    const float* __restrict__ bias, float* __restrict__ out,
                    int H, int W, float negslope, const float* __restrict__ residual)
{
    __shared__ __bf16 wlds[9 * 32 * 32];   // [tap][n][k]
    __shared__ __bf16 tile[3 * 34 * 32];   // [row][x][c], x padded by 1 each side

    const int tid = threadIdx.x;
    const int b = blockIdx.z, y = blockIdx.y;
    const int xb0 = blockIdx.x * 32;

    // weights: [n][k][tap] -> [tap][n][k]
    for (int i = tid; i < 9 * 1024; i += 128) {
        int k = i & 31, n = (i >> 5) & 31, tap = i >> 10;
        wlds[i] = (__bf16)w[(n * 32 + k) * 9 + tap];
    }
    // input tile: rows y-1..y+1, x in [xb0-1, xb0+32], zero-padded OOB
    for (int i = tid; i < 3 * 32 * 34; i += 128) {
        int xi = i % 34;
        int rc = i / 34;
        int c = rc & 31, r = rc >> 5;
        int yy = y + r - 1, xx = xb0 + xi - 1;
        float v = (yy >= 0 && yy < H && xx >= 0 && xx < W)
                      ? in[((b * 32 + c) * H + yy) * W + xx] : 0.f;
        tile[(r * 34 + xi) * 32 + c] = (__bf16)v;
    }
    // prefetch next row's tile (global_prefetch_b8)
    if (y + 2 < H) {
        int xx = xb0 + (tid & 31);
        if (xx < W)
            __builtin_prefetch(&in[((b * 32 + (tid >> 2) % 32) * H + (y + 2)) * W + xx], 0, 3);
    }
    __syncthreads();

    const int wave = tid >> 5, lane = tid & 31;
    const int mbase = (wave & 1) * 16;
    const int n0 = (wave >> 1) * 16;
    const int m = lane & 15, half = lane >> 4;

    v8f acc = {};
    #pragma unroll
    for (int tap = 0; tap < 9; ++tap) {
        const int ky = tap / 3, kx = tap % 3;
        const __bf16* ap = &tile[(ky * 34 + mbase + m + kx) * 32 + half * 8];
        v8bf lo = *(const v8bf*)ap;          // channels half*8 .. +7
        v8bf hi = *(const v8bf*)(ap + 16);   // channels 16+half*8 .. +7
        v16bf a = cat8(lo, hi);
        const __bf16* bp = &wlds[(tap * 32 + n0 + (lane & 15)) * 32 + half * 8];
        v16bf bb = cat8(*(const v8bf*)bp, *(const v8bf*)(bp + 16));
        acc = wmma_bf16(a, bb, acc);
    }

    const int nl = n0 + (lane & 15);
    const float bv = bias[nl];
    #pragma unroll
    for (int r = 0; r < 8; ++r) {
        const int xo = mbase + xb0 + r + 8 * half - (wave & 1) * 16 + (wave & 1) * 16;
        (void)xo;
    }
    #pragma unroll
    for (int r = 0; r < 8; ++r) {
        const int xo = xb0 + mbase + r + 8 * half;
        if (xo < W) {
            float v = acc[r] + bv;
            if (negslope > 0.0f) v = (v >= 0.0f) ? v : negslope * v;
            const int oidx = ((b * 32 + nl) * H + y) * W + xo;
            if (residual) v += residual[oidx];
            out[oidx] = v;
        }
    }
}

// ---------------------------------------------------------------------------
// conv1x1 (GEMM N=32,K=32) via WMMA bf16, fused const-add / residual / nan_to_num
// grid: x = HW/32, y = B ; block = 128 (4 waves)
// ---------------------------------------------------------------------------
__global__ __launch_bounds__(128)
void k_conv1x1_wmma(const float* __restrict__ in, const float* __restrict__ w,
                    const float* __restrict__ bias, float* __restrict__ out,
                    int HW, float cadd, const float* __restrict__ residual, int nanfix)
{
    __shared__ __bf16 wlds[32 * 32];   // [n][k] (same as global layout)
    __shared__ __bf16 tile[32 * 32];   // [px][c]

    const int tid = threadIdx.x;
    const int b = blockIdx.y;
    const int p0blk = blockIdx.x * 32;

    for (int i = tid; i < 1024; i += 128)
        wlds[i] = (__bf16)w[i];
    for (int i = tid; i < 1024; i += 128) {
        int c = i >> 5, px = i & 31;                     // px fast -> coalesced
        tile[px * 32 + c] = (__bf16)in[(b * 32 + c) * HW + p0blk + px];
    }
    __syncthreads();

    const int wave = tid >> 5, lane = tid & 31;
    const int mbase = (wave & 1) * 16;
    const int n0 = (wave >> 1) * 16;
    const int m = lane & 15, half = lane >> 4;

    const __bf16* ap = &tile[(mbase + m) * 32 + half * 8];
    v16bf a = cat8(*(const v8bf*)ap, *(const v8bf*)(ap + 16));
    const __bf16* bp = &wlds[(n0 + (lane & 15)) * 32 + half * 8];
    v16bf bb = cat8(*(const v8bf*)bp, *(const v8bf*)(bp + 16));

    v8f acc = {};
    acc = wmma_bf16(a, bb, acc);

    const int nl = n0 + (lane & 15);
    const float bv = bias[nl];
    #pragma unroll
    for (int r = 0; r < 8; ++r) {
        const int pix = p0blk + mbase + r + 8 * half;
        float v = acc[r] + bv + cadd;
        const int oidx = (b * 32 + nl) * HW + pix;
        if (residual) v += residual[oidx];
        if (nanfix && !isfinite(v)) v = 1e-5f;
        out[oidx] = v;
    }
}

// ---------------------------------------------------------------------------
// InstanceNorm statistics: one block per (b,c); 256 threads reduce H*W elems
// ---------------------------------------------------------------------------
__global__ __launch_bounds__(256)
void k_inorm_stats(const float* __restrict__ x, float* __restrict__ mu,
                   float* __restrict__ istd, int HW)
{
    const int bc = blockIdx.x;
    const float* p = x + (long)bc * HW;
    float s = 0.f, s2 = 0.f;
    for (int i = threadIdx.x; i < HW; i += 256) {
        float v = p[i];
        s += v; s2 += v * v;
    }
    __shared__ float sh[256], sh2[256];
    sh[threadIdx.x] = s; sh2[threadIdx.x] = s2;
    __syncthreads();
    for (int o = 128; o > 0; o >>= 1) {
        if (threadIdx.x < o) { sh[threadIdx.x] += sh[threadIdx.x + o];
                               sh2[threadIdx.x] += sh2[threadIdx.x + o]; }
        __syncthreads();
    }
    if (threadIdx.x == 0) {
        float mean = sh[0] / (float)HW;
        float var  = sh2[0] / (float)HW - mean * mean;
        mu[bc] = mean;
        istd[bc] = rsqrtf(var + 1e-5f);
    }
}

// ---------------------------------------------------------------------------
// exchange-mask thresholds: 1 block of 32 threads
// ---------------------------------------------------------------------------
__global__ void k_thresh(const float* __restrict__ g0, const float* __restrict__ g1,
                         float* __restrict__ mask1, float* __restrict__ mask2)
{
    __shared__ float sa[32];
    __shared__ float thr;
    const int c = threadIdx.x;

    float a0 = fabsf(g0[c]);
    sa[c] = a0; __syncthreads();
    if (c == 0) {
        float mn = sa[0], mx = sa[0];
        for (int i = 1; i < 32; ++i) { mn = fminf(mn, sa[i]); mx = fmaxf(mx, sa[i]); }
        thr = mn + 0.05f * (mx - mn);
    }
    __syncthreads();
    mask1[c] = (a0 >= thr) ? 1.f : 0.f;
    __syncthreads();

    float a1 = fabsf(g1[c]);
    sa[c] = a1; __syncthreads();
    if (c == 0) {
        float mn = sa[0], mx = sa[0];
        for (int i = 1; i < 32; ++i) { mn = fminf(mn, sa[i]); mx = fmaxf(mx, sa[i]); }
        thr = mn + 0.05f * (mx - mn);
    }
    __syncthreads();
    mask2[c] = (a1 >= thr) ? 1.f : 0.f;
}

// ---------------------------------------------------------------------------
// fused instance-norm apply + channel exchange -> X8 = [o1 ; o2] (batch 8)
// HW = 65536 so bc = idx>>16
// ---------------------------------------------------------------------------
__global__ __launch_bounds__(256)
void k_exchange(const float* __restrict__ t1, const float* __restrict__ t2,
                const float* __restrict__ mu1, const float* __restrict__ is1,
                const float* __restrict__ mu2, const float* __restrict__ is2,
                const float* __restrict__ g0, const float* __restrict__ b0,
                const float* __restrict__ g1, const float* __restrict__ b1,
                const float* __restrict__ m1, const float* __restrict__ m2,
                float* __restrict__ X8, int N)
{
    int idx = blockIdx.x * 256 + threadIdx.x;
    if (idx >= N) return;
    const int bc = idx >> 16;       // HW = 65536
    const int c  = bc & 31;
    float n1 = (t1[idx] - mu1[bc]) * is1[bc] * g0[c] + b0[c];
    float n2 = (t2[idx] - mu2[bc]) * is2[bc] * g1[c] + b1[c];
    float o1 = n1 + ((m1[c] != 0.f) ? 0.f : n2);
    float o2 = n2 + ((m2[c] != 0.f) ? 0.f : n1);
    X8[idx]     = o1;
    X8[idx + N] = o2;
}

// ---------------------------------------------------------------------------
// 256-point Stockham radix-2 FFT (LDS ping-pong, 128 threads)
// sign = -1 forward, +1 inverse
// ---------------------------------------------------------------------------
static __device__ __forceinline__ void fft256(float2* __restrict__ bufA,
                                              float2* __restrict__ bufB,
                                              int t, float sign, float2** resOut)
{
    float2* src = bufA;
    float2* dst = bufB;
    #pragma unroll
    for (int s = 0; s < 8; ++s) {
        const int m = 1 << s;
        const int l = 128 >> s;
        const int j = t >> s;
        float2 c0 = src[t];
        float2 c1 = src[t + 128];
        float ang = sign * M_PI_F * (float)j / (float)l;
        float wr = cosf(ang), wi = sinf(ang);
        float2 sum = make_float2(c0.x + c1.x, c0.y + c1.y);
        float2 dif = make_float2(c0.x - c1.x, c0.y - c1.y);
        float2 tw  = make_float2(dif.x * wr - dif.y * wi, dif.x * wi + dif.y * wr);
        const int jm = j << s;
        dst[t + jm]     = sum;
        dst[t + jm + m] = tw;
        __syncthreads();
        float2* tmp = src; src = dst; dst = tmp;
    }
    *resOut = src;   // after 8 swaps: bufA
}

// rows rFFT: one block per row of 256 reals -> 129 complex bins
__global__ __launch_bounds__(128)
void k_fft_rows(const float* __restrict__ in, float2* __restrict__ out)
{
    __shared__ float2 bufA[256], bufB[256];
    const int row = blockIdx.x;
    const int t = threadIdx.x;
    bufA[t]       = make_float2(in[row * 256 + t], 0.f);
    bufA[t + 128] = make_float2(in[row * 256 + t + 128], 0.f);
    __syncthreads();
    float2* res;
    fft256(bufA, bufB, t, -1.f, &res);
    out[row * 129 + t] = res[t];
    if (t == 0) out[row * 129 + 128] = res[128];
}

// cols FFT + amp/phase: one block per (img, wbin)
__global__ __launch_bounds__(128)
void k_fft_cols(const float2* __restrict__ in, float* __restrict__ amp,
                float* __restrict__ pha)
{
    __shared__ float2 bufA[256], bufB[256];
    const int wb  = blockIdx.x;           // 0..128
    const int img = blockIdx.y;           // 0..255
    const int t = threadIdx.x;
    const long base = (long)img * 256 * 129 + wb;
    bufA[t]       = in[base + (long)t * 129];
    bufA[t + 128] = in[base + (long)(t + 128) * 129];
    __syncthreads();
    float2* res;
    fft256(bufA, bufB, t, -1.f, &res);
    #pragma unroll
    for (int h = 0; h < 2; ++h) {
        const int y = t + h * 128;
        float2 v = res[y];
        amp[base + (long)y * 129] = sqrtf(v.x * v.x + v.y * v.y);
        pha[base + (long)y * 129] = atan2f(v.y, v.x);
    }
}

// inverse cols FFT, fused z = (af*cos(pf)+2e-8) + i(af*sin(pf)+1e-8)
__global__ __launch_bounds__(128)
void k_ifft_cols(const float* __restrict__ af, const float* __restrict__ pf,
                 float2* __restrict__ out)
{
    __shared__ float2 bufA[256], bufB[256];
    const int wb  = blockIdx.x;
    const int img = blockIdx.y;
    const int t = threadIdx.x;
    const long base = (long)img * 256 * 129 + wb;
    #pragma unroll
    for (int h = 0; h < 2; ++h) {
        const int y = t + h * 128;
        float a = af[base + (long)y * 129];
        float p = pf[base + (long)y * 129];
        bufA[y] = make_float2(a * cosf(p) + 2e-8f, a * sinf(p) + 1e-8f);
    }
    __syncthreads();
    float2* res;
    fft256(bufA, bufB, t, +1.f, &res);
    out[base + (long)t * 129]         = res[t];
    out[base + (long)(t + 128) * 129] = res[t + 128];
}

// inverse rows (irfft): Hermitian rebuild -> 256-pt inverse -> |real/(H*W)|
__global__ __launch_bounds__(128)
void k_irfft_rows(const float2* __restrict__ in, float* __restrict__ out)
{
    __shared__ float2 bufA[256], bufB[256];
    const int row = blockIdx.x;
    const int t = threadIdx.x;
    float2 v = in[row * 129 + t];
    bufA[t] = v;
    if (t == 0) bufA[128] = in[row * 129 + 128];
    else        bufA[256 - t] = make_float2(v.x, -v.y);
    __syncthreads();
    float2* res;
    fft256(bufA, bufB, t, +1.f, &res);
    const float nrm = 1.f / 65536.f;
    out[row * 256 + t]       = fabsf(res[t].x * nrm);
    out[row * 256 + t + 128] = fabsf(res[t + 128].x * nrm);
}

// ---------------------------------------------------------------------------
// host-side orchestration
// ---------------------------------------------------------------------------
extern "C" void kernel_launch(void* const* d_in, const int* in_sizes, int n_in,
                              void* d_out, int out_size, void* d_ws, size_t ws_size,
                              hipStream_t stream)
{
    (void)in_sizes; (void)n_in; (void)out_size; (void)ws_size;
    const float* x1     = (const float*)d_in[0];
    const float* x2     = (const float*)d_in[1];
    const float* conv_w = (const float*)d_in[2];
    const float* conv_b = (const float*)d_in[3];
    const float* in0_w  = (const float*)d_in[4];
    const float* in0_b  = (const float*)d_in[5];
    const float* in1_w  = (const float*)d_in[6];
    const float* in1_b  = (const float*)d_in[7];
    const float* fpre_w = (const float*)d_in[8];
    const float* fpre_b = (const float*)d_in[9];
    const float* amp1_w = (const float*)d_in[10];
    const float* amp1_b = (const float*)d_in[11];
    const float* amp2_w = (const float*)d_in[12];
    const float* amp2_b = (const float*)d_in[13];
    const float* pha1_w = (const float*)d_in[14];
    const float* pha1_b = (const float*)d_in[15];
    const float* pha2_w = (const float*)d_in[16];
    const float* pha2_b = (const float*)d_in[17];
    const float* post_w = (const float*)d_in[18];
    const float* post_b = (const float*)d_in[19];
    float* outp = (float*)d_out;

    const int Bh = 4, C = 32, H = 256, W = 256;
    const int HW = H * W;                       // 65536
    const int NCHW = Bh * C * HW;               // 8,388,608
    const int Ws = W / 2 + 1;                   // 129
    const int SPEC = 8 * C * H * Ws;            // 8,454,144

    float* ws = (float*)d_ws;
    float* mu1   = ws;        float* is1 = ws + 128;
    float* mu2   = ws + 256;  float* is2 = ws + 384;
    float* mask1 = ws + 512;  float* mask2 = ws + 544;
    size_t off = 1024;
    float* t1 = ws + off; off += (size_t)NCHW;
    float* t2 = ws + off; off += (size_t)NCHW;
    float* X8 = ws + off; off += (size_t)2 * NCHW;
    float* F0 = ws + off; off += (size_t)2 * NCHW;      // fpre out, reused as R
    float* C1 = ws + off; off += (size_t)2 * SPEC;      // complex, reused as C2
    float* A0 = ws + off; off += (size_t)SPEC;
    float* P0 = ws + off; off += (size_t)SPEC;
    float* TMP = ws + off; off += (size_t)SPEC;
    float* AF = ws + off; off += (size_t)SPEC;
    float* PF = ws + off; off += (size_t)SPEC;

    // 1) ExchangeBlock convs
    dim3 gconv(W / 32, H, Bh);
    k_conv3x3_wmma<<<gconv, 128, 0, stream>>>(x1, conv_w, conv_b, t1, H, W, 0.f, nullptr);
    k_conv3x3_wmma<<<gconv, 128, 0, stream>>>(x2, conv_w, conv_b, t2, H, W, 0.f, nullptr);

    // 2) instance-norm stats + masks
    k_inorm_stats<<<Bh * C, 256, 0, stream>>>(t1, mu1, is1, HW);
    k_inorm_stats<<<Bh * C, 256, 0, stream>>>(t2, mu2, is2, HW);
    k_thresh<<<1, 32, 0, stream>>>(in0_w, in1_w, mask1, mask2);

    // 3) fused norm-apply + exchange -> X8 (batch 8, also the post residual)
    k_exchange<<<(NCHW + 255) / 256, 256, 0, stream>>>(
        t1, t2, mu1, is1, mu2, is2, in0_w, in0_b, in1_w, in1_b, mask1, mask2, X8, NCHW);

    // 4) fpre 1x1 conv + 1e-8
    k_conv1x1_wmma<<<dim3(HW / 32, 8), 128, 0, stream>>>(
        X8, fpre_w, fpre_b, F0, HW, 1e-8f, nullptr, 0);

    // 5) rfft2 = row rFFT then column FFT (amp/phase fused)
    k_fft_rows<<<8 * C * H, 128, 0, stream>>>(F0, (float2*)C1);
    k_fft_cols<<<dim3(Ws, 8 * C), 128, 0, stream>>>((float2*)C1, A0, P0);

    // 6) amp / phase conv chains on the 256x129 spectrum
    dim3 gspec((Ws + 31) / 32, H, 8);
    k_conv3x3_wmma<<<gspec, 128, 0, stream>>>(A0, amp1_w, amp1_b, TMP, H, Ws, 0.1f, nullptr);
    k_conv3x3_wmma<<<gspec, 128, 0, stream>>>(TMP, amp2_w, amp2_b, AF, H, Ws, 0.f, A0);
    k_conv3x3_wmma<<<gspec, 128, 0, stream>>>(P0, pha1_w, pha1_b, TMP, H, Ws, 0.1f, nullptr);
    k_conv3x3_wmma<<<gspec, 128, 0, stream>>>(TMP, pha2_w, pha2_b, PF, H, Ws, 0.f, P0);

    // 7) irfft2: fused recombine + inverse column FFT, then inverse row rFFT + |.|
    k_ifft_cols<<<dim3(Ws, 8 * C), 128, 0, stream>>>(AF, PF, (float2*)C1);
    k_irfft_rows<<<8 * C * H, 128, 0, stream>>>((float2*)C1, F0);

    // 8) post 1x1 conv + residual + nan_to_num -> d_out ([2,B,C,H,W])
    k_conv1x1_wmma<<<dim3(HW / 32, 8), 128, 0, stream>>>(
        F0, post_w, post_b, outp, HW, 0.f, X8, 1);
}
